// Encoder_74775380623962
// MI455X (gfx1250) — matrix operand order repaired and checked by reading, not compiled, not run on hardware
//
#include <hip/hip_runtime.h>
#include <hip/hip_bf16.h>

typedef __attribute__((ext_vector_type(16))) __bf16 v16bf;
typedef __attribute__((ext_vector_type(8)))  float  v8f;

#define N0   200000
#define N1   50000
#define N2   10000
#define E0   800000
#define E1   160000
#define F_IN 128
#define HID  256
#define L_OUT 32

__device__ __forceinline__ unsigned short f2bf(float f) {
    unsigned u = __float_as_uint(f);
    unsigned r = u + 0x7FFFu + ((u >> 16) & 1u);   // round-to-nearest-even
    return (unsigned short)(r >> 16);
}
__device__ __forceinline__ float bf2f(unsigned short s) {
    return __uint_as_float(((unsigned)s) << 16);
}
__device__ __forceinline__ __bf16 bfbits(unsigned short s) {
    union { unsigned short u; __bf16 b; } c; c.u = s; return c.b;
}

// ---------------- h0 = bf16(log(x+1)) ----------------
__global__ void k_log1p(const float* __restrict__ x, unsigned short* __restrict__ h0, int total) {
    int i = blockIdx.x * blockDim.x + threadIdx.x;
    if (i < total) h0[i] = f2bf(__logf(x[i] + 1.0f));
}

__global__ void k_zero(float* __restrict__ p, int total) {
    int i = blockIdx.x * blockDim.x + threadIdx.x;
    if (i < total) p[i] = 0.0f;
}

// ---- pack fp32 KxN weight into per-lane WMMA B-fragment layout (bf16) ----
// out[((kt*NT+nt)*32 + lane)*16 + e] = W[kt*32 + 16*(lane>>4) + e][nt*16 + (lane&15)]
__global__ void k_pack(const float* __restrict__ W, unsigned short* __restrict__ out, int K, int N) {
    int i = blockIdx.x * blockDim.x + threadIdx.x;
    if (i >= K * N) return;
    int e    = i & 15;
    int lane = (i >> 4) & 31;
    int tile = i >> 9;
    int NT   = N >> 4;
    int nt   = tile % NT;
    int kt   = tile / NT;
    int k = kt * 32 + ((lane >> 4) << 4) + e;
    int n = nt * 16 + (lane & 15);
    out[i] = f2bf(W[k * N + n]);
}

// ---------------- edge scatter: agg[dst] += h[src]; cnt[dst] += 1 ----------------
__global__ void k_scatter(const unsigned short* __restrict__ h,
                          const int* __restrict__ src, const int* __restrict__ dst,
                          float* __restrict__ agg, float* __restrict__ cnt,
                          int nedges, int flog2) {
    int F = 1 << flog2;
    int idx = blockIdx.x * blockDim.x + threadIdx.x;
    if (idx >= (nedges << flog2)) return;
    int e = idx >> flog2;
    int f = idx & (F - 1);
    int s = src[e], d = dst[e];
    atomicAdd(&agg[d * F + f], bf2f(h[s * F + f]));
    if (f == 0) atomicAdd(&cnt[d], 1.0f);
}

// A-fragment K index for 16-bit WMMA (contiguous, matches k_pack):
// k = kt*32 + 16*(lane>>4) + e
__device__ __forceinline__ int afrag_k(int kt, int e, int half) {
    return kt * 32 + (half << 4) + e;
}

// ---------------- layer 0: dual GEMM (K=128,N=256) + bias + relu + L2 norm ----------------
__global__ void __launch_bounds__(256) k_layer0(
    const unsigned short* __restrict__ h0, const float* __restrict__ agg0, const float* __restrict__ cnt0,
    const unsigned short* __restrict__ pWs, const unsigned short* __restrict__ pWn,
    const float* __restrict__ bias, unsigned short* __restrict__ h1)
{
    __shared__ unsigned short sA[16 * 128];
    __shared__ unsigned short sG[16 * 128];
    __shared__ float rowsq[16];
    const int tid  = threadIdx.x;
    const int row0 = blockIdx.x * 16;

    for (int idx = tid; idx < 16 * 128; idx += 256) {
        int r = idx >> 7, f = idx & 127;
        int node = row0 + r;
        sA[idx] = h0[node * 128 + f];
        float c = fmaxf(cnt0[node], 1.0f);
        sG[idx] = f2bf(agg0[node * 128 + f] / c);
    }
    if (tid < 16) rowsq[tid] = 0.0f;
    __syncthreads();

    const int wave = tid >> 5, lane = tid & 31;
    const int half = lane >> 4, lm = lane & 15;
    const int NT = HID / 16;

    v8f acc[2] = { {0.f,0.f,0.f,0.f,0.f,0.f,0.f,0.f},
                   {0.f,0.f,0.f,0.f,0.f,0.f,0.f,0.f} };

    #pragma unroll
    for (int kt = 0; kt < 4; kt++) {
        v16bf aS, aG;
        #pragma unroll
        for (int e = 0; e < 16; e++) {
            int k = afrag_k(kt, e, half);
            aS[e] = bfbits(sA[lm * 128 + k]);
            aG[e] = bfbits(sG[lm * 128 + k]);
        }
        #pragma unroll
        for (int t = 0; t < 2; t++) {
            int nt = wave * 2 + t;
            v16bf bS = *reinterpret_cast<const v16bf*>(pWs + ((kt * NT + nt) * 32 + lane) * 16);
            v16bf bN = *reinterpret_cast<const v16bf*>(pWn + ((kt * NT + nt) * 32 + lane) * 16);
            acc[t] = __builtin_amdgcn_wmma_f32_16x16x32_bf16(false, aS, false, bS, (short)0, acc[t], false, false);
            acc[t] = __builtin_amdgcn_wmma_f32_16x16x32_bf16(false, aG, false, bN, (short)0, acc[t], false, false);
        }
    }

    float vals[2][8];
    #pragma unroll
    for (int t = 0; t < 2; t++) {
        int n = (wave * 2 + t) * 16 + lm;
        float b = bias[n];
        #pragma unroll
        for (int r = 0; r < 8; r++) {
            float v = fmaxf(acc[t][r] + b, 0.0f);
            vals[t][r] = v;
            atomicAdd(&rowsq[r + half * 8], v * v);
        }
    }
    __syncthreads();

    #pragma unroll
    for (int t = 0; t < 2; t++) {
        int n = (wave * 2 + t) * 16 + lm;
        #pragma unroll
        for (int r = 0; r < 8; r++) {
            int row = r + half * 8;
            float inv = 1.0f / fmaxf(sqrtf(rowsq[row]), 1e-12f);
            h1[(row0 + row) * HID + n] = f2bf(vals[t][r] * inv);
        }
    }
}

// ---------------- layer 1 (K=256,N=256) + fused heads (K=256,N=32 each) ----------------
__global__ void __launch_bounds__(256) k_layer1_head(
    const unsigned short* __restrict__ h1, const float* __restrict__ agg1, const float* __restrict__ cnt1,
    const unsigned short* __restrict__ pWs, const unsigned short* __restrict__ pWn,
    const float* __restrict__ b1,
    const unsigned short* __restrict__ pWmu, const float* __restrict__ bmu,
    const unsigned short* __restrict__ pWvar, const float* __restrict__ bvar,
    float* __restrict__ zloc, float* __restrict__ zscale)
{
    __shared__ unsigned short sA[16 * 256];
    __shared__ unsigned short sG[16 * 256];
    __shared__ unsigned short sH[16 * 256];
    __shared__ float rowsq[16];
    const int tid  = threadIdx.x;
    const int row0 = blockIdx.x * 16;

    for (int idx = tid; idx < 16 * 256; idx += 256) {
        int r = idx >> 8, f = idx & 255;
        int node = row0 + r;
        sA[idx] = h1[node * 256 + f];
        float c = fmaxf(cnt1[node], 1.0f);
        sG[idx] = f2bf(agg1[node * 256 + f] / c);
    }
    if (tid < 16) rowsq[tid] = 0.0f;
    __syncthreads();

    const int wave = tid >> 5, lane = tid & 31;
    const int half = lane >> 4, lm = lane & 15;
    const int NT = HID / 16;

    v8f acc[2] = { {0.f,0.f,0.f,0.f,0.f,0.f,0.f,0.f},
                   {0.f,0.f,0.f,0.f,0.f,0.f,0.f,0.f} };

    #pragma unroll
    for (int kt = 0; kt < 8; kt++) {
        v16bf aS, aG;
        #pragma unroll
        for (int e = 0; e < 16; e++) {
            int k = afrag_k(kt, e, half);
            aS[e] = bfbits(sA[lm * 256 + k]);
            aG[e] = bfbits(sG[lm * 256 + k]);
        }
        #pragma unroll
        for (int t = 0; t < 2; t++) {
            int nt = wave * 2 + t;
            v16bf bS = *reinterpret_cast<const v16bf*>(pWs + ((kt * NT + nt) * 32 + lane) * 16);
            v16bf bN = *reinterpret_cast<const v16bf*>(pWn + ((kt * NT + nt) * 32 + lane) * 16);
            acc[t] = __builtin_amdgcn_wmma_f32_16x16x32_bf16(false, aS, false, bS, (short)0, acc[t], false, false);
            acc[t] = __builtin_amdgcn_wmma_f32_16x16x32_bf16(false, aG, false, bN, (short)0, acc[t], false, false);
        }
    }

    float vals[2][8];
    #pragma unroll
    for (int t = 0; t < 2; t++) {
        int n = (wave * 2 + t) * 16 + lm;
        float b = b1[n];
        #pragma unroll
        for (int r = 0; r < 8; r++) {
            float v = fmaxf(acc[t][r] + b, 0.0f);
            vals[t][r] = v;
            atomicAdd(&rowsq[r + half * 8], v * v);
        }
    }
    __syncthreads();

    #pragma unroll
    for (int t = 0; t < 2; t++) {
        int n = (wave * 2 + t) * 16 + lm;
        #pragma unroll
        for (int r = 0; r < 8; r++) {
            int row = r + half * 8;
            float inv = 1.0f / fmaxf(sqrtf(rowsq[row]), 1e-12f);
            sH[row * 256 + n] = f2bf(vals[t][r] * inv);
        }
    }
    __syncthreads();

    // Heads: wave0/1 -> z_loc cols 0-15 / 16-31 ; wave2/3 -> z_scale
    if (wave < 4) {
        const unsigned short* pW = (wave < 2) ? pWmu : pWvar;
        int nt = wave & 1;
        v8f a2 = {0.f,0.f,0.f,0.f,0.f,0.f,0.f,0.f};
        #pragma unroll
        for (int kt = 0; kt < 8; kt++) {
            v16bf a;
            #pragma unroll
            for (int e = 0; e < 16; e++) {
                int k = afrag_k(kt, e, half);
                a[e] = bfbits(sH[lm * 256 + k]);
            }
            v16bf b = *reinterpret_cast<const v16bf*>(pW + ((kt * 2 + nt) * 32 + lane) * 16);
            a2 = __builtin_amdgcn_wmma_f32_16x16x32_bf16(false, a, false, b, (short)0, a2, false, false);
        }
        int col = nt * 16 + lm;
        if (wave < 2) {
            float bb = bmu[col];
            #pragma unroll
            for (int r = 0; r < 8; r++)
                zloc[(row0 + r + half * 8) * L_OUT + col] = a2[r] + bb;
        } else {
            float bb = bvar[col];
            #pragma unroll
            for (int r = 0; r < 8; r++)
                zscale[(row0 + r + half * 8) * L_OUT + col] = expf(a2[r] + bb) + 1e-6f;
        }
    }
}

extern "C" void kernel_launch(void* const* d_in, const int* in_sizes, int n_in,
                              void* d_out, int out_size, void* d_ws, size_t ws_size,
                              hipStream_t stream)
{
    const float* x    = (const float*)d_in[0];
    const float* Ws0  = (const float*)d_in[1];
    const float* Wn0  = (const float*)d_in[2];
    const float* b0   = (const float*)d_in[3];
    const float* Ws1  = (const float*)d_in[4];
    const float* Wn1  = (const float*)d_in[5];
    const float* b1   = (const float*)d_in[6];
    const float* Wmu  = (const float*)d_in[7];
    const float* bmu  = (const float*)d_in[8];
    const float* Wvar = (const float*)d_in[9];
    const float* bvar = (const float*)d_in[10];
    const int* src0 = (const int*)d_in[11];
    const int* dst0 = (const int*)d_in[12];
    const int* src1 = (const int*)d_in[13];
    const int* dst1 = (const int*)d_in[14];

    char* ws = (char*)d_ws;
    size_t off = 0;
    auto alloc = [&](size_t bytes) -> void* {
        void* p = ws + off;
        off = (off + bytes + 255) & ~(size_t)255;
        return p;
    };
    unsigned short* h0   = (unsigned short*)alloc((size_t)N0 * F_IN * 2);
    unsigned short* h1   = (unsigned short*)alloc((size_t)N1 * HID * 2);
    float* agg0 = (float*)alloc((size_t)N1 * F_IN * 4);
    float* cnt0 = (float*)alloc((size_t)N1 * 4);
    float* agg1 = (float*)alloc((size_t)N2 * HID * 4);
    float* cnt1 = (float*)alloc((size_t)N2 * 4);
    unsigned short* pWs0  = (unsigned short*)alloc((size_t)F_IN * HID * 2);
    unsigned short* pWn0  = (unsigned short*)alloc((size_t)F_IN * HID * 2);
    unsigned short* pWs1  = (unsigned short*)alloc((size_t)HID * HID * 2);
    unsigned short* pWn1  = (unsigned short*)alloc((size_t)HID * HID * 2);
    unsigned short* pWmu  = (unsigned short*)alloc((size_t)HID * L_OUT * 2);
    unsigned short* pWvar = (unsigned short*)alloc((size_t)HID * L_OUT * 2);

    float* zloc   = (float*)d_out;
    float* zscale = zloc + (size_t)N2 * L_OUT;

    const int B = 256;
    // 1) h0 = bf16(log1p(x))  -- gather table becomes L2-resident (51 MB < 192 MB L2)
    k_log1p<<<(N0 * F_IN + B - 1) / B, B, 0, stream>>>(x, h0, N0 * F_IN);
    // 2) zero accumulators (must re-zero every call)
    k_zero<<<(N1 * F_IN + B - 1) / B, B, 0, stream>>>(agg0, N1 * F_IN);
    k_zero<<<(N1 + B - 1) / B, B, 0, stream>>>(cnt0, N1);
    k_zero<<<(N2 * HID + B - 1) / B, B, 0, stream>>>(agg1, N2 * HID);
    k_zero<<<(N2 + B - 1) / B, B, 0, stream>>>(cnt1, N2);
    // 3) pack weights into WMMA B-fragment layout (bf16)
    k_pack<<<(F_IN * HID + B - 1) / B, B, 0, stream>>>(Ws0, pWs0, F_IN, HID);
    k_pack<<<(F_IN * HID + B - 1) / B, B, 0, stream>>>(Wn0, pWn0, F_IN, HID);
    k_pack<<<(HID * HID + B - 1) / B, B, 0, stream>>>(Ws1, pWs1, HID, HID);
    k_pack<<<(HID * HID + B - 1) / B, B, 0, stream>>>(Wn1, pWn1, HID, HID);
    k_pack<<<(HID * L_OUT + B - 1) / B, B, 0, stream>>>(Wmu, pWmu, HID, L_OUT);
    k_pack<<<(HID * L_OUT + B - 1) / B, B, 0, stream>>>(Wvar, pWvar, HID, L_OUT);
    // 4) layer 0 aggregate + GEMM
    k_scatter<<<(E0 * F_IN + B - 1) / B, B, 0, stream>>>(h0, src0, dst0, agg0, cnt0, E0, 7);
    k_layer0<<<N1 / 16, B, 0, stream>>>(h0, agg0, cnt0, pWs0, pWn0, b0, h1);
    // 5) layer 1 aggregate + GEMM + fused heads
    k_scatter<<<(E1 * HID + B - 1) / B, B, 0, stream>>>(h1, src1, dst1, agg1, cnt1, E1, 8);
    k_layer1_head<<<N2 / 16, B, 0, stream>>>(h1, agg1, cnt1, pWs1, pWn1, b1,
                                             pWmu, bmu, pWvar, bvar, zloc, zscale);
}
